// LSTM_80161269613040
// MI455X (gfx1250) — compile-verified, matrix operand order checked by reading, not run on hardware
//
#include <hip/hip_runtime.h>
#include <hip/hip_bf16.h>
#include <math.h>

// ---------------------------------------------------------------------------
// LSTM forward, N=32, T=2048, D=512, H=512  (MI455X / gfx1250, wave32)
//   prep  : pack Wx/Wh into bf16 WMMA-B fragment layout; x -> bf16
//   phase1: gates(bf16) = x @ Wx   (bf16 WMMA, fp32 accum, 16x32 per wave)
//   phase2: persistent 32-WG scan; Wh slice in LDS; h in bf16; gate tiles
//           prefetched with GLOBAL_LOAD_ASYNC_TO_LDS_B128 double buffering;
//           c/bias in registers; grid barrier per step.
// ---------------------------------------------------------------------------

typedef __attribute__((ext_vector_type(16))) __bf16 v16bf;
typedef __attribute__((ext_vector_type(8)))  __bf16 v8bf;
typedef __attribute__((ext_vector_type(8)))  float  v8f;
typedef __attribute__((ext_vector_type(4)))  float  f32x4;

#define N_   32
#define T_   2048
#define D_   512
#define H_   512
#define G_   2048            // 4*H
#define NWG  32              // persistent workgroups in the scan
#define CPW  16              // h-columns owned per workgroup
#define KT_W 16              // 512 / 32 K-tiles
#define CT_W 128             // 2048 / 16 column tiles

__device__ __forceinline__ int kmap(int lane, int e) {
  return ((lane & 16) ? 8 : 0) + (e < 8 ? e : e + 8);
}

// A fragment from a bf16 row-major row: two aligned b128 loads, no converts.
__device__ __forceinline__ v16bf load_a_bf(const __bf16* rowbase, int kb) {
  v8bf lo = *(const v8bf*)(rowbase + kb);
  v8bf hi = *(const v8bf*)(rowbase + kb + 16);
  v16bf a;
#pragma unroll
  for (int i = 0; i < 8; ++i) { a[i] = lo[i]; a[8 + i] = hi[i]; }
  return a;
}

// A fragment from an fp32 row (minimal-workspace fallback only).
__device__ __forceinline__ v16bf load_a_f32(const float* rowbase, int kb) {
  const f32x4* p0 = (const f32x4*)(rowbase + kb);
  const f32x4* p1 = (const f32x4*)(rowbase + kb + 16);
  f32x4 f0 = p0[0], f1 = p0[1], f2 = p1[0], f3 = p1[1];
  v16bf a;
#pragma unroll
  for (int i = 0; i < 4; ++i) {
    a[i]      = (__bf16)f0[i];
    a[4 + i]  = (__bf16)f1[i];
    a[8 + i]  = (__bf16)f2[i];
    a[12 + i] = (__bf16)f3[i];
  }
  return a;
}

// Packed B fragment (global or LDS): 512 bf16 per block, 32B/lane contiguous.
__device__ __forceinline__ v16bf load_b_pk(const __bf16* packed, int blk, int lane) {
  return *(const v16bf*)(packed + (size_t)blk * 512 + lane * 16);
}

__device__ __forceinline__ float sigmoidf_(float x) { return 1.f / (1.f + __expf(-x)); }
__device__ __forceinline__ float tanhf_(float x)    { return 1.f - 2.f / (__expf(2.f * x) + 1.f); }

// CDNA5 async gather into LDS (tracked by ASYNCcnt). lds_off is the wave-
// relative LDS byte address (low 32 bits of a generic shared pointer).
__device__ __forceinline__ void async_ld_lds_b128(unsigned lds_off,
                                                  unsigned long long gaddr) {
  asm volatile("global_load_async_to_lds_b128 %0, %1, off"
               :: "v"(lds_off), "v"(gaddr) : "memory");
}
__device__ __forceinline__ void wait_asynccnt_le1() {
  asm volatile("s_wait_asynccnt 0x1" ::: "memory");
}

// ---------------------------------------------------------------------------
// Pack (512 x 2048) fp32 row-major weights into bf16 B-fragment blocks.
// ---------------------------------------------------------------------------
__global__ void pack_w(const float* __restrict__ W, __bf16* __restrict__ P) {
  int i = blockIdx.x * blockDim.x + threadIdx.x;
  if (i >= D_ * G_) return;
  int e    = i & 15;
  int lane = (i >> 4) & 31;
  int blk  = i >> 9;
  int kt   = blk >> 7;
  int ct   = blk & 127;
  int K    = kt * 32 + kmap(lane, e);
  int col  = ct * 16 + (lane & 15);
  P[i] = (__bf16)W[(size_t)K * G_ + col];
}

// x (fp32) -> bf16, same row-major layout.
__global__ void cvt_x(const float* __restrict__ x, __bf16* __restrict__ xb) {
  size_t i = (size_t)blockIdx.x * blockDim.x + threadIdx.x;
  if (i < (size_t)N_ * T_ * D_) xb[i] = (__bf16)x[i];
}

// h0 (fp32) -> bf16 hbuf[0]; reset grid barrier state.
__global__ void init_state(const float* __restrict__ h0, __bf16* __restrict__ hbuf,
                           unsigned* __restrict__ bar) {
  int i = blockIdx.x * blockDim.x + threadIdx.x;
  if (i < N_ * H_) hbuf[i] = (__bf16)h0[i];
  if (i < 4) bar[i] = 0u;
}

// ---------------------------------------------------------------------------
// Phase 1: gates = x @ Wx (bf16 out). Each wave: a 16x32 output tile.
// ---------------------------------------------------------------------------
__global__ void __launch_bounds__(256)
gates_gemm(const __bf16* __restrict__ xb, const __bf16* __restrict__ pWx,
           __bf16* __restrict__ gates) {
  const int wave = threadIdx.x >> 5, lane = threadIdx.x & 31;
  const int tile = blockIdx.x * 8 + wave;
  const int mt   = tile >> 6;
  const int ct0  = (tile & 63) * 2;
  const int kbase = (lane & 16) ? 8 : 0;
  const __bf16* arow = xb + (size_t)(mt * 16 + (lane & 15)) * D_;

  v8f acc0 = {}, acc1 = {};
#pragma unroll 4
  for (int kt = 0; kt < KT_W; ++kt) {
    v16bf a  = load_a_bf(arow, kt * 32 + kbase);
    v16bf b0 = load_b_pk(pWx, kt * CT_W + ct0, lane);
    v16bf b1 = load_b_pk(pWx, kt * CT_W + ct0 + 1, lane);
    acc0 = __builtin_amdgcn_wmma_f32_16x16x32_bf16(false, a, false, b0,
                                                   (short)0, acc0, false, false);
    acc1 = __builtin_amdgcn_wmma_f32_16x16x32_bf16(false, a, false, b1,
                                                   (short)0, acc1, false, false);
  }
  const int chi = (lane & 16) ? 8 : 0;
  const int c0  = ct0 * 16 + (lane & 15);
#pragma unroll
  for (int v = 0; v < 8; ++v) {
    size_t r = (size_t)(mt * 16 + chi + v);
    gates[r * G_ + c0]      = (__bf16)acc0[v];
    gates[r * G_ + c0 + 16] = (__bf16)acc1[v];
  }
}

// ---------------------------------------------------------------------------
// Phase 2: persistent scan. MODE: 0 = gates precomputed (async-prefetched),
// 1 = fused (x bf16), 2 = fused (x fp32, minimal workspace).
// Dynamic LDS: [ Wh 64K | (Wx 64K if fused) | a 8K | (gate pf 2x4K if MODE 0) ]
// ---------------------------------------------------------------------------
template <int MODE>
__global__ void __launch_bounds__(256)
lstm_scan(const void* __restrict__ xAny, const __bf16* __restrict__ pWx,
          const __bf16* __restrict__ pWh, const float* __restrict__ bias,
          const __bf16* __restrict__ gates, __bf16* __restrict__ hbuf,
          unsigned* __restrict__ bar, float* __restrict__ out) {
  constexpr int NSTAGE = (MODE == 0) ? 1 : 2;
  extern __shared__ char smem[];
  __bf16* sW    = (__bf16*)smem;                       // NSTAGE * 32768 bf16
  float*  a_lds = (float*)(smem + NSTAGE * 65536);     // [4][N_][CPW] = 8 KB
  __bf16* pf    = (__bf16*)(smem + NSTAGE * 65536 + 8192); // 2 x 2048 bf16

  const int wg   = blockIdx.x;
  const int tid  = threadIdx.x;
  const int wave = tid >> 5, lane = tid & 31;
  const int q = wave & 3;               // gate index for this wave's tile
  const int m = wave >> 2;              // M tile (rows m*16 .. m*16+15)
  const int kbase = (lane & 16) ? 8 : 0;
  const int chi   = (lane & 16) ? 8 : 0;
  const int col0  = wg * CPW;

  // ---- stage this WG's weight slices into LDS, blk = q*16+kt
  {
    uint4* dst = (uint4*)sW;
    for (int i = tid; i < NSTAGE * 4096; i += 256) {
      int half = i >> 12;               // 0 = Wh, 1 = Wx
      int j    = i & 4095;
      int blk  = j >> 6, within = j & 63;
      int qq = blk >> 4, kt = blk & 15;
      const __bf16* src = (half ? pWx : pWh) +
                          (size_t)(kt * CT_W + qq * 32 + wg) * 512;
      dst[i] = ((const uint4*)src)[within];
    }
  }

  // ---- per-thread fixed ownership: elements (r0,cc) and (r1,cc)
  const int cc = tid & 15;
  const int r0 = tid >> 4;              // 0..15
  const int r1 = r0 + 16;               // 16..31
  float bb[4];
#pragma unroll
  for (int g = 0; g < 4; ++g) bb[g] = bias[g * H_ + col0 + cc];
  float cA = 0.f, cB = 0.f;             // cell state in registers

  // ---- async gate prefetch helper state (MODE 0)
  const unsigned pfBase = (unsigned)(uintptr_t)(void*)pf;
  const int pj = tid & 7, pr = tid >> 3;        // 8 x 16B segments per row
  const int pg = pj >> 1, ph = pj & 1;
  auto issue_pf = [&](int tt, int bufSel) {
    const __bf16* gp = gates + ((size_t)pr * T_ + tt) * G_ +
                       pg * H_ + col0 + ph * 8;
    unsigned loff = pfBase + bufSel * 4096 + pr * 128 + pj * 16;
    async_ld_lds_b128(loff, (unsigned long long)(uintptr_t)gp);
  };
  if (MODE == 0) issue_pf(0, 0);
  __syncthreads();

  const __bf16* sWh = sW;
  const __bf16* sWx = sW + 32768;

  int p = 0;
  for (int t = 0; t < T_; ++t) {
    if (MODE == 0) {                    // prefetch t+1 while t computes
      int tn = (t + 1 < T_) ? (t + 1) : (T_ - 1);
      issue_pf(tn, (t + 1) & 1);
    }
    const __bf16* hb = hbuf + (size_t)p * (N_ * H_);

    // ---- recurrent GEMM tile: h @ Wh[:, gate tile], B from LDS
    v8f acc = {};
    {
      const __bf16* arow = hb + (size_t)(m * 16 + (lane & 15)) * H_;
#pragma unroll 4
      for (int kt = 0; kt < KT_W; ++kt) {
        v16bf a = load_a_bf(arow, kt * 32 + kbase);
        v16bf b = load_b_pk(sWh, q * 16 + kt, lane);
        acc = __builtin_amdgcn_wmma_f32_16x16x32_bf16(false, a, false, b,
                                                      (short)0, acc, false, false);
      }
    }
    if (MODE == 1) {
      const __bf16* xrow = (const __bf16*)xAny +
                           ((size_t)(m * 16 + (lane & 15)) * T_ + t) * D_;
#pragma unroll 4
      for (int kt = 0; kt < KT_W; ++kt) {
        v16bf a = load_a_bf(xrow, kt * 32 + kbase);
        v16bf b = load_b_pk(sWx, q * 16 + kt, lane);
        acc = __builtin_amdgcn_wmma_f32_16x16x32_bf16(false, a, false, b,
                                                      (short)0, acc, false, false);
      }
    } else if (MODE == 2) {
      const float* xrow = (const float*)xAny +
                          ((size_t)(m * 16 + (lane & 15)) * T_ + t) * D_;
#pragma unroll 4
      for (int kt = 0; kt < KT_W; ++kt) {
        v16bf a = load_a_f32(xrow, kt * 32 + kbase);
        v16bf b = load_b_pk(sWx, q * 16 + kt, lane);
        acc = __builtin_amdgcn_wmma_f32_16x16x32_bf16(false, a, false, b,
                                                      (short)0, acc, false, false);
      }
    }

    // ---- scatter fp32 accumulator tile to LDS for the gate fusion
#pragma unroll
    for (int v = 0; v < 8; ++v)
      a_lds[(q * N_ + (m * 16 + chi + v)) * CPW + (lane & 15)] = acc[v];
    if (MODE == 0) wait_asynccnt_le1();   // t's gate tile resident in LDS
    __syncthreads();

    // ---- fused LSTM cell update (two fixed elements per thread)
    __bf16* hnext = hbuf + (size_t)(p ^ 1) * (N_ * H_);
    const __bf16* pft = pf + (t & 1) * 2048;
#pragma unroll
    for (int piece = 0; piece < 2; ++piece) {
      int r = piece ? r1 : r0;
      float av[4];
#pragma unroll
      for (int g = 0; g < 4; ++g) {
        av[g] = a_lds[(g * N_ + r) * CPW + cc] + bb[g];
        if (MODE == 0) av[g] += (float)pft[r * 64 + g * 16 + cc];
      }
      float ig = sigmoidf_(av[0]);
      float fg = sigmoidf_(av[1]);
      float og = sigmoidf_(av[2]);
      float gg = tanhf_(av[3]);
      float& cr = piece ? cB : cA;
      cr = fg * cr + ig * gg;
      float hn = og * tanhf_(cr);
      hnext[(size_t)r * H_ + col0 + cc] = (__bf16)hn;
      out[((size_t)r * T_ + t) * H_ + col0 + cc] = hn;
    }

    // ---- device-wide sense-reversal barrier
    __threadfence();
    __syncthreads();
    if (tid == 0) {
      unsigned gen = __hip_atomic_load(&bar[1], __ATOMIC_ACQUIRE,
                                       __HIP_MEMORY_SCOPE_AGENT);
      unsigned arr = __hip_atomic_fetch_add(&bar[0], 1u, __ATOMIC_ACQ_REL,
                                            __HIP_MEMORY_SCOPE_AGENT);
      if (arr == NWG - 1) {
        __hip_atomic_store(&bar[0], 0u, __ATOMIC_RELAXED,
                           __HIP_MEMORY_SCOPE_AGENT);
        __hip_atomic_fetch_add(&bar[1], 1u, __ATOMIC_RELEASE,
                               __HIP_MEMORY_SCOPE_AGENT);
      } else {
        while (__hip_atomic_load(&bar[1], __ATOMIC_ACQUIRE,
                                 __HIP_MEMORY_SCOPE_AGENT) == gen) {
          __builtin_amdgcn_s_sleep(1);
        }
      }
    }
    __syncthreads();
    p ^= 1;
  }
}

// ---------------------------------------------------------------------------
extern "C" void kernel_launch(void* const* d_in, const int* in_sizes, int n_in,
                              void* d_out, int out_size, void* d_ws, size_t ws_size,
                              hipStream_t stream) {
  const float* x  = (const float*)d_in[0];  // (N, T, D)
  const float* h0 = (const float*)d_in[1];  // (N, H)
  const float* Wx = (const float*)d_in[2];  // (D, 4H)
  const float* Wh = (const float*)d_in[3];  // (H, 4H)
  const float* b  = (const float*)d_in[4];  // (4H,)
  float* out = (float*)d_out;               // (N, T, H)

  char* ws = (char*)d_ws;
  size_t off = 0;
  auto take = [&](size_t bytes) {
    char* p = ws + off;
    off = (off + bytes + 255) & ~(size_t)255;
    return p;
  };
  __bf16*   pWx  = (__bf16*)take((size_t)D_ * G_ * sizeof(__bf16));    // 2 MB
  __bf16*   pWh  = (__bf16*)take((size_t)D_ * G_ * sizeof(__bf16));    // 2 MB
  __bf16*   hbuf = (__bf16*)take((size_t)2 * N_ * H_ * sizeof(__bf16));// 64 KB
  unsigned* bar  = (unsigned*)take(256);
  const size_t baseOff    = off;
  const size_t xbBytes    = (size_t)N_ * T_ * D_ * sizeof(__bf16);     // 64 MB
  const size_t gatesBytes = (size_t)N_ * T_ * G_ * sizeof(__bf16);     // 256 MB

  int mode;                     // 0 precomp, 1 fused bf16-x, 2 fused f32-x
  __bf16* xb    = nullptr;
  __bf16* gates = nullptr;
  if (ws_size >= baseOff + xbBytes + gatesBytes) {
    xb    = (__bf16*)take(xbBytes);
    gates = (__bf16*)take(gatesBytes);
    mode = 0;
  } else if (ws_size >= baseOff + xbBytes) {
    xb = (__bf16*)take(xbBytes);
    mode = 1;
  } else {
    mode = 2;
  }

  const int packElems = D_ * G_;
  pack_w<<<(packElems + 255) / 256, 256, 0, stream>>>(Wx, pWx);
  pack_w<<<(packElems + 255) / 256, 256, 0, stream>>>(Wh, pWh);
  init_state<<<(N_ * H_ + 255) / 256, 256, 0, stream>>>(h0, hbuf, bar);
  if (mode <= 1) {
    const size_t xElems = (size_t)N_ * T_ * D_;
    cvt_x<<<(unsigned)((xElems + 255) / 256), 256, 0, stream>>>(x, xb);
  }

  const size_t shPrecomp = 65536 + 8192 + 8192;       // Wh + a + gate pf
  const size_t shFused   = 2 * 65536 + 8192 + 8192;   // Wh + Wx + a (+ pad)

  if (mode == 0) {
    const int tiles = (N_ * T_ / 16) * (G_ / 32);     // 16x32 output per wave
    gates_gemm<<<tiles / 8, 256, 0, stream>>>(xb, pWx, gates);
    lstm_scan<0><<<NWG, 256, shPrecomp, stream>>>(nullptr, pWx, pWh, b, gates,
                                                  hbuf, bar, out);
  } else if (mode == 1) {
    lstm_scan<1><<<NWG, 256, shFused, stream>>>(xb, pWx, pWh, b, nullptr,
                                                hbuf, bar, out);
  } else {
    lstm_scan<2><<<NWG, 256, shFused, stream>>>(x, pWx, pWh, b, nullptr,
                                                hbuf, bar, out);
  }
}
